// ScaleDotProductAttention_43611097924069
// MI455X (gfx1250) — matrix operand order repaired and checked by reading, not compiled, not run on hardware
//
#include <hip/hip_runtime.h>

typedef _Float16 v8h  __attribute__((ext_vector_type(8)));
typedef _Float16 v16h __attribute__((ext_vector_type(16)));
typedef float    v8f  __attribute__((ext_vector_type(8)));

#define B_  2
#define H_  16
#define L_  2048
#define D_  64
#define NEGV (-1000000000.0f)
#define SCALE 0.125f            /* 1/sqrt(64) */
#define ROWS_PER_BLOCK 128
#define KT_TILE 32              /* keys per flash iteration */

__device__ __forceinline__ v8h load8f_to_h(const float* p) {
    const float4* p4 = (const float4*)p;
    float4 a = p4[0];
    float4 b = p4[1];
    v8h r;
    r[0] = (_Float16)a.x; r[1] = (_Float16)a.y; r[2] = (_Float16)a.z; r[3] = (_Float16)a.w;
    r[4] = (_Float16)b.x; r[5] = (_Float16)b.y; r[6] = (_Float16)b.z; r[7] = (_Float16)b.w;
    return r;
}

__device__ __forceinline__ v16h cat16(v8h lo, v8h hi) {
    return __builtin_shufflevector(lo, hi, 0,1,2,3,4,5,6,7,8,9,10,11,12,13,14,15);
}

__device__ __forceinline__ v8h lds8(const _Float16* p) {
    return *(const v8h*)p;        /* 16B-aligned by construction -> ds_load_b128 */
}

__global__ __launch_bounds__(256)
void sdpa_flash_wmma_kernel(const float* __restrict__ q,
                            const float* __restrict__ k,
                            const float* __restrict__ v,
                            const int*   __restrict__ mask,
                            float*       __restrict__ out)
{
    /* K-tile transposed [e_local][d], V-tile transposed [d][e_local], per-wave P staging */
    __shared__ _Float16 KT[32][72];     /* rows 144B (16B mult); reads at d-offsets 0/16/32/48 */
    __shared__ _Float16 VT[64][48];     /* rows  96B (16B mult); reads at e-offsets 0/8/16/24 */
    __shared__ _Float16 PT[8][16][32];  /* rows  64B */

    const int t    = threadIdx.x;
    const int wave = t >> 5;
    const int lane = t & 31;
    const int lh   = lane >> 4;   /* lane half: 0 or 1 */
    const int ln   = lane & 15;

    const int bh   = blockIdx.y;
    const int b    = bh >> 4;                          /* / H_ */
    const int row0 = blockIdx.x * ROWS_PER_BLOCK + wave * 16;

    const float* qb = q + (size_t)bh * L_ * D_;
    const float* kb = k + (size_t)bh * L_ * D_;        /* viewed row-major as [64][2048] */
    const float* vb = v + (size_t)bh * L_ * D_;
    const int*   mb = mask + (size_t)b * L_ * L_;
    float*       ob = out + (size_t)bh * L_ * D_;

    /* ---- load this wave's Q tile (16x64) into two f16 A-fragments ---- */
    const float* qrow = qb + (size_t)(row0 + ln) * D_;
    v16h aq0, aq1;
    {
        const int c0 = lh * 16;  /* per-lane K chunk base inside each 32-wide group */
        v8h lo = load8f_to_h(qrow + c0);
        v8h hi = load8f_to_h(qrow + c0 + 8);
        aq0 = cat16(lo, hi);                /* covers d = 0..31  */
        lo = load8f_to_h(qrow + 32 + c0);
        hi = load8f_to_h(qrow + 32 + c0 + 8);
        aq1 = cat16(lo, hi);                /* covers d = 32..63 */
    }

    v8f o0 = {}, o1 = {}, o2 = {}, o3 = {};
    float mrow[8], lrow[8];
#pragma unroll
    for (int r = 0; r < 8; ++r) { mrow[r] = -3.0e38f; lrow[r] = 0.0f; }

    for (int e0 = 0; e0 < L_; e0 += KT_TILE) {
        __syncthreads();
        /* stage K~ tile transposed: KT[e][d] = kflat[d*2048 + e0+e]  (coalesced reads) */
#pragma unroll
        for (int i = 0; i < 8; ++i) {
            int idx = i * 256 + t;
            int d = idx >> 5, e = idx & 31;
            KT[e][d] = (_Float16)kb[(size_t)d * L_ + e0 + e];
        }
        /* stage V tile transposed: VT[d][e] = v[(e0+e)*64 + d]  (coalesced reads) */
#pragma unroll
        for (int i = 0; i < 8; ++i) {
            int idx = i * 256 + t;
            int e = idx >> 6, d = idx & 63;
            VT[d][e] = (_Float16)vb[(size_t)(e0 + e) * D_ + d];
        }
        __syncthreads();

        /* ---- S(16x32) = Q(16x64) x K~(64x32), two 16x16 N-tiles ---- */
        v8f s0 = {}, s1 = {};
        {
            v16h bk = cat16(lds8(&KT[ln][lh * 16]),      lds8(&KT[ln][lh * 16 + 8]));
            s0 = __builtin_amdgcn_wmma_f32_16x16x32_f16(false, aq0, false, bk, (short)0, s0, false, false);
            bk = cat16(lds8(&KT[ln][32 + lh * 16]),      lds8(&KT[ln][32 + lh * 16 + 8]));
            s0 = __builtin_amdgcn_wmma_f32_16x16x32_f16(false, aq1, false, bk, (short)0, s0, false, false);
            bk = cat16(lds8(&KT[16 + ln][lh * 16]),      lds8(&KT[16 + ln][lh * 16 + 8]));
            s1 = __builtin_amdgcn_wmma_f32_16x16x32_f16(false, aq0, false, bk, (short)0, s1, false, false);
            bk = cat16(lds8(&KT[16 + ln][32 + lh * 16]), lds8(&KT[16 + ln][32 + lh * 16 + 8]));
            s1 = __builtin_amdgcn_wmma_f32_16x16x32_f16(false, aq1, false, bk, (short)0, s1, false, false);
        }

        /* ---- mask + online softmax (rows live across the 16 lanes of each half) ---- */
#pragma unroll
        for (int r = 0; r < 8; ++r) {
            const int grow = row0 + r + lh * 8;
            const int* mp = mb + (size_t)grow * L_ + e0;
            float sc0 = s0[r] * SCALE;
            float sc1 = s1[r] * SCALE;
            if (mp[ln] == 0)      sc0 = NEGV;
            if (mp[16 + ln] == 0) sc1 = NEGV;
            float mx = fmaxf(sc0, sc1);
            mx = fmaxf(mx, __shfl_xor(mx, 1));
            mx = fmaxf(mx, __shfl_xor(mx, 2));
            mx = fmaxf(mx, __shfl_xor(mx, 4));
            mx = fmaxf(mx, __shfl_xor(mx, 8));
            float mnew  = fmaxf(mrow[r], mx);
            float alpha = __expf(mrow[r] - mnew);
            float p0 = __expf(sc0 - mnew);
            float p1 = __expf(sc1 - mnew);
            float rs = p0 + p1;
            rs += __shfl_xor(rs, 1);
            rs += __shfl_xor(rs, 2);
            rs += __shfl_xor(rs, 4);
            rs += __shfl_xor(rs, 8);
            lrow[r] = lrow[r] * alpha + rs;
            mrow[r] = mnew;
            o0[r] *= alpha; o1[r] *= alpha; o2[r] *= alpha; o3[r] *= alpha;
            const int rl = r + lh * 8;
            PT[wave][rl][ln]      = (_Float16)p0;   /* C-layout -> LDS (row-major P) */
            PT[wave][rl][16 + ln] = (_Float16)p1;
        }
        asm volatile("s_wait_dscnt 0" ::: "memory");  /* same-wave LDS store->load */

        /* ---- O(16x64) += P(16x32) x V(32x64), four 16x16 N-tiles ---- */
        v16h ap = cat16(lds8(&PT[wave][ln][lh * 16]),
                        lds8(&PT[wave][ln][lh * 16 + 8]));
        v16h bv;
        bv = cat16(lds8(&VT[ln][lh * 16]),      lds8(&VT[ln][lh * 16 + 8]));
        o0 = __builtin_amdgcn_wmma_f32_16x16x32_f16(false, ap, false, bv, (short)0, o0, false, false);
        bv = cat16(lds8(&VT[16 + ln][lh * 16]), lds8(&VT[16 + ln][lh * 16 + 8]));
        o1 = __builtin_amdgcn_wmma_f32_16x16x32_f16(false, ap, false, bv, (short)0, o1, false, false);
        bv = cat16(lds8(&VT[32 + ln][lh * 16]), lds8(&VT[32 + ln][lh * 16 + 8]));
        o2 = __builtin_amdgcn_wmma_f32_16x16x32_f16(false, ap, false, bv, (short)0, o2, false, false);
        bv = cat16(lds8(&VT[48 + ln][lh * 16]), lds8(&VT[48 + ln][lh * 16 + 8]));
        o3 = __builtin_amdgcn_wmma_f32_16x16x32_f16(false, ap, false, bv, (short)0, o3, false, false);
    }

    /* ---- epilogue: normalize by row sums and store fp32 ---- */
#pragma unroll
    for (int r = 0; r < 8; ++r) {
        const float inv = 1.0f / lrow[r];
        const int grow = row0 + r + lh * 8;
        float* orow = ob + (size_t)grow * D_;
        orow[ln]      = o0[r] * inv;
        orow[16 + ln] = o1[r] * inv;
        orow[32 + ln] = o2[r] * inv;
        orow[48 + ln] = o3[r] * inv;
    }
}

extern "C" void kernel_launch(void* const* d_in, const int* in_sizes, int n_in,
                              void* d_out, int out_size, void* d_ws, size_t ws_size,
                              hipStream_t stream) {
    (void)in_sizes; (void)n_in; (void)out_size; (void)d_ws; (void)ws_size;
    const float* q    = (const float*)d_in[0];
    const float* k    = (const float*)d_in[1];
    const float* v    = (const float*)d_in[2];
    const int*   mask = (const int*)d_in[3];
    float* out = (float*)d_out;

    dim3 grid(L_ / ROWS_PER_BLOCK, B_ * H_);  /* 16 x 32 = 512 workgroups */
    sdpa_flash_wmma_kernel<<<grid, 256, 0, stream>>>(q, k, v, mask, out);
}
